// DynamicConv2d_37830071943906
// MI455X (gfx1250) — matrix operand order repaired and checked by reading, not compiled, not run on hardware
//
#include <hip/hip_runtime.h>

typedef __attribute__((ext_vector_type(8)))  __bf16 v8bf;
typedef __attribute__((ext_vector_type(16))) __bf16 v16bf;
typedef __attribute__((ext_vector_type(8)))  float  v8f;

typedef __attribute__((address_space(1))) int* gptr_i32;
typedef __attribute__((address_space(3))) int* lptr_i32;

#define CIN   128
#define COUT  128
#define HW    96
#define KTAPS 9
#define CPAD  136            // cin padding: 408 bf16 c-stride -> bank-conflict-free b128

#if __has_builtin(__builtin_amdgcn_global_load_async_to_lds_b32)
#define HAS_ASYNC_LDS 1
#else
#define HAS_ASYNC_LDS 0
#endif

// ---------------------------------------------------------------------------
// Kernel 1: aggregate per-sample weights, split fp32 -> (hi,lo) bf16, repack
// to [b][tap][cout][cin] (cin innermost) to match the 16-bit WMMA A-fragment
// layout (per-lane contiguous K runs along cin).
// ---------------------------------------------------------------------------
__global__ __launch_bounds__(256) void aggw_pack_kernel(
    const float* __restrict__ attn,     // [16,4]
    const float* __restrict__ weight,   // [4,128,128,3,3]
    __bf16* __restrict__ whi,           // [16,9,128cout,128cin]
    __bf16* __restrict__ wlo)           // [16,9,128cout,128cin]
{
    int tid = blockIdx.x * 256 + threadIdx.x;      // 16*9*128*128 = 2359296
    int i = tid & 127;                 // cin  (innermost -> coalesced writes)
    int o = (tid >> 7) & 127;          // cout
    int t = (tid >> 14) % 9;           // tap
    int b = tid / (9 << 14);
    if (b >= 16) return;
    int kh = t / 3, kw = t % 3;

    const size_t kstride = (size_t)COUT * CIN * KTAPS;   // 147456
    size_t widx = (((size_t)o * CIN + i) * 3 + kh) * 3 + kw;

    float s = 0.0f;
#pragma unroll
    for (int k = 0; k < 4; ++k)
        s += attn[b * 4 + k] * weight[widx + (size_t)k * kstride];

    __bf16 hi = (__bf16)s;
    __bf16 lo = (__bf16)(s - (float)hi);

    size_t oidx = (((size_t)b * KTAPS + t) * COUT + o) * CIN + i;
    whi[oidx] = hi;
    wlo[oidx] = lo;
}

// ---------------------------------------------------------------------------
// Kernel 2: implicit-GEMM dynamic conv, bf16x3 split on V_WMMA_F32_16X16X32_BF16.
// Block: 256 thr / 8 waves = (batch b, row h, 32-pixel chunk).
// Wave w: couts [16w,16w+16) x 32 pixels (two 16x16 C tiles).
// Staging: GLOBAL_LOAD_ASYNC_TO_LDS (ASYNCcnt) for the halo tile, then an
// LDS->LDS fp32 -> (hi,lo) bf16 split pass.
// ---------------------------------------------------------------------------
__global__ __launch_bounds__(256) void dynconv_wmma_kernel(
    const float* __restrict__ x,        // [16,128,96,96]
    const float* __restrict__ attn,     // [16,4]
    const float* __restrict__ biasp,    // [4,128]
    const __bf16* __restrict__ whi,     // [16,9,128,128]
    const __bf16* __restrict__ wlo,     // [16,9,128,128]
    float* __restrict__ out)            // [16,128,96,96]
{
    // split tile: xs[c:34][r:3][cin:CPAD]; c covers col0-1..col0+32,
    // r covers rows h-1..h+1 (zero-padded at image borders).
    __shared__ __align__(16) __bf16 xs_hi[34 * 3 * CPAD];   // 27744 B
    __shared__ __align__(16) __bf16 xs_lo[34 * 3 * CPAD];   // 27744 B
#if HAS_ASYNC_LDS
    __shared__ __align__(16) float  xraw[CIN * 3 * 34];     // 52224 B raw fp32 tile
#endif

    const int cchunk = blockIdx.x;       // 0..2
    const int h      = blockIdx.y;       // 0..95
    const int b      = blockIdx.z;       // 0..15
    const int tid    = threadIdx.x;
    const int col0   = cchunk * 32;

    const float* xb = x + (size_t)b * CIN * HW * HW;

#if HAS_ASYNC_LDS
    // ---- phase 1: async-copy halo tile global->LDS (EXEC-predicated per lane;
    //      out-of-image slots get disjoint ds_store zeros -> no ordering hazard)
    for (int idx = tid; idx < CIN * 3 * 34; idx += 256) {
        int c = idx % 34;
        int r = (idx / 34) % 3;
        int i = idx / (34 * 3);
        int hh = h - 1 + r;
        int cc = col0 - 1 + c;
        if ((unsigned)hh < (unsigned)HW && (unsigned)cc < (unsigned)HW) {
            float* gp = const_cast<float*>(xb + ((size_t)i * HW + hh) * HW + cc);
            __builtin_amdgcn_global_load_async_to_lds_b32(
                (gptr_i32)gp, (lptr_i32)&xraw[idx], 0, 0);
        } else {
            xraw[idx] = 0.0f;
        }
    }
#if __has_builtin(__builtin_amdgcn_s_wait_asynccnt)
    __builtin_amdgcn_s_wait_asynccnt(0);
#else
    asm volatile("s_wait_asynccnt 0x0" ::: "memory");
#endif
    __syncthreads();

    // ---- phase 2: LDS->LDS split fp32 -> (hi, lo) bf16
    for (int idx = tid; idx < CIN * 3 * 34; idx += 256) {
        int c = idx % 34;
        int r = (idx / 34) % 3;
        int i = idx / (34 * 3);
        float v = xraw[idx];
        __bf16 hi = (__bf16)v;
        __bf16 lo = (__bf16)(v - (float)hi);
        int soff = (c * 3 + r) * CPAD + i;
        xs_hi[soff] = hi;
        xs_lo[soff] = lo;
    }
#else
    // ---- fallback staging: direct global read + split
    for (int idx = tid; idx < CIN * 3 * 34; idx += 256) {
        int c = idx % 34;
        int r = (idx / 34) % 3;
        int i = idx / (34 * 3);
        int hh = h - 1 + r;
        int cc = col0 - 1 + c;
        float v = 0.0f;
        if ((unsigned)hh < (unsigned)HW && (unsigned)cc < (unsigned)HW)
            v = xb[((size_t)i * HW + hh) * HW + cc];
        __bf16 hi = (__bf16)v;
        __bf16 lo = (__bf16)(v - (float)hi);
        int soff = (c * 3 + r) * CPAD + i;
        xs_hi[soff] = hi;
        xs_lo[soff] = lo;
    }
#endif
    __syncthreads();

    const int wave = tid >> 5;
    const int lane = tid & 31;
    const int half = lane >> 4;          // selects K sub-runs per 16-bit layout
    const int ln   = lane & 15;          // M for A, N for B/C/D
    const int coutBase = wave * 16;

    v8f acc0 = {};
    v8f acc1 = {};

    const size_t bbase = (size_t)b * KTAPS;

    for (int t9 = 0; t9 < KTAPS; ++t9) {
        const int kh = t9 / 3, kw = t9 % 3;
        // A: per-lane row cout = coutBase+ln, cin innermost
        const size_t wrow = ((bbase + t9) * COUT + coutBase + ln) * (size_t)CIN;
        const __bf16* wha = whi + wrow;
        const __bf16* wla = wlo + wrow;
        // B: per-lane pixel column; tiles at n and n+16
        const int soff0 = ((ln +      kw) * 3 + kh) * CPAD;
        const int soff1 = ((ln + 16 + kw) * 3 + kh) * CPAD;

#pragma unroll
        for (int i0 = 0; i0 < CIN; i0 += 32) {
            // ---- A fragments 16x32: half-wave K=0..15 / 16..31, contiguous cin
            const int ka = i0 + half * 16;
            v8bf ah0 = *(const v8bf*)__builtin_assume_aligned(wha + ka, 16);
            v8bf ah1 = *(const v8bf*)__builtin_assume_aligned(wha + ka + 8, 16);
            v8bf al0 = *(const v8bf*)__builtin_assume_aligned(wla + ka, 16);
            v8bf al1 = *(const v8bf*)__builtin_assume_aligned(wla + ka + 8, 16);
            v16bf Ah = __builtin_shufflevector(ah0, ah1, 0,1,2,3,4,5,6,7,8,9,10,11,12,13,14,15);
            v16bf Al = __builtin_shufflevector(al0, al1, 0,1,2,3,4,5,6,7,8,9,10,11,12,13,14,15);

            // ---- B fragments 32x16: half-wave K=0..15 / 16..31, contiguous cin
            const int kb = i0 + half * 16;
            v16bf B0h = *(const v16bf*)__builtin_assume_aligned(xs_hi + soff0 + kb, 16);
            v16bf B0l = *(const v16bf*)__builtin_assume_aligned(xs_lo + soff0 + kb, 16);
            v16bf B1h = *(const v16bf*)__builtin_assume_aligned(xs_hi + soff1 + kb, 16);
            v16bf B1l = *(const v16bf*)__builtin_assume_aligned(xs_lo + soff1 + kb, 16);

            // ---- bf16x3: hi*hi + hi*lo + lo*hi  (lo*lo ~2^-16, dropped)
            acc0 = __builtin_amdgcn_wmma_f32_16x16x32_bf16(false, Ah, false, B0h, (short)0, acc0, false, false);
            acc0 = __builtin_amdgcn_wmma_f32_16x16x32_bf16(false, Ah, false, B0l, (short)0, acc0, false, false);
            acc0 = __builtin_amdgcn_wmma_f32_16x16x32_bf16(false, Al, false, B0h, (short)0, acc0, false, false);
            acc1 = __builtin_amdgcn_wmma_f32_16x16x32_bf16(false, Ah, false, B1h, (short)0, acc1, false, false);
            acc1 = __builtin_amdgcn_wmma_f32_16x16x32_bf16(false, Ah, false, B1l, (short)0, acc1, false, false);
            acc1 = __builtin_amdgcn_wmma_f32_16x16x32_bf16(false, Al, false, B1h, (short)0, acc1, false, false);
        }
    }

    // ---- epilogue: exact-fp32 aggregated bias, store two 16x16 tiles
    const float a0 = attn[b * 4 + 0], a1 = attn[b * 4 + 1];
    const float a2 = attn[b * 4 + 2], a3 = attn[b * 4 + 3];
    float* ob = out + (size_t)b * COUT * HW * HW + (size_t)h * HW;

#pragma unroll
    for (int v = 0; v < 8; ++v) {
        const int cout = coutBase + half * 8 + v;   // C/D: M = v + 8*half, N = ln
        const float bias = a0 * biasp[cout]            + a1 * biasp[COUT + cout]
                         + a2 * biasp[2 * COUT + cout] + a3 * biasp[3 * COUT + cout];
        float* op = ob + (size_t)cout * HW * HW + col0 + ln;
        op[0]  = acc0[v] + bias;
        op[16] = acc1[v] + bias;
    }
}

// ---------------------------------------------------------------------------
extern "C" void kernel_launch(void* const* d_in, const int* in_sizes, int n_in,
                              void* d_out, int out_size, void* d_ws, size_t ws_size,
                              hipStream_t stream)
{
    const float* x      = (const float*)d_in[0];   // [16,128,96,96]
    const float* attn   = (const float*)d_in[1];   // [16,4]
    const float* weight = (const float*)d_in[2];   // [4,128,128,3,3]
    const float* biasp  = (const float*)d_in[3];   // [4,128]
    float* out = (float*)d_out;

    const size_t nW = (size_t)16 * KTAPS * COUT * CIN;     // 2359296
    __bf16* whi = (__bf16*)d_ws;                           // 4.72 MB
    __bf16* wlo = whi + nW;                                // 4.72 MB (9.44 MB total)

    aggw_pack_kernel<<<9216, 256, 0, stream>>>(attn, weight, whi, wlo);

    dim3 grid(3, HW, 16);
    dynconv_wmma_kernel<<<grid, 256, 0, stream>>>(x, attn, biasp, whi, wlo, out);
}
